// MultiHeadAttention_1477468750006
// MI455X (gfx1250) — matrix operand order repaired and checked by reading, not compiled, not run on hardware
//
#include <hip/hip_runtime.h>

// ---------------------------------------------------------------------------
// MHA forward for MI455X (gfx1250): bf16 WMMA everywhere, flash attention,
// double-buffered TDM (tensor_load_to_lds) staging of K/V tiles.
// Fragment layouts used here (k contiguous per lane half, same order in A and B):
//   A 16x32 bf16 frag : lanes 0-15 = row, hold K[k0+0..15];
//                       lanes 16-31 hold K[k0+16..31]
//   B 32x16 bf16 frag : lanes 0-15 = col, hold K[k0+0..15]; lanes 16-31 K[k0+16..31]
//   C/D 16x16 f32     : VGPR g, lanes 0-15 -> (M=g, N=lane); lanes 16-31 -> (M=8+g)
// ---------------------------------------------------------------------------

typedef __attribute__((ext_vector_type(16))) __bf16 v16bf;
typedef __attribute__((ext_vector_type(8)))  float  v8f;

#define DM     512   // d_model
#define S_LEN  4096
#define NH     8
#define DHEAD  64

__device__ __forceinline__ unsigned short f2bf(float f) {
  unsigned int u = __float_as_uint(f);
  u += 0x7FFFu + ((u >> 16) & 1u);          // round-to-nearest-even
  return (unsigned short)(u >> 16);
}

union Frag32B { v16bf v; uint4 q[2]; unsigned short s[16]; };

__device__ __forceinline__ v8f zero8() {
  v8f z;
#pragma unroll
  for (int i = 0; i < 8; ++i) z[i] = 0.0f;
  return z;
}

// A-fragment from a row-major bf16 row pointer, K window at k0, per-lane half.
__device__ __forceinline__ v16bf load_afrag(const unsigned short* row, int k0, int half) {
  Frag32B f;
  f.q[0] = *(const uint4*)(row + k0 + half * 16);       // K[k0 + 16h .. +7]
  f.q[1] = *(const uint4*)(row + k0 + half * 16 + 8);   // K[k0+16h+8 .. +7]
  return f.v;
}

// A-fragment built from fp32 memory (converted to bf16 on the fly).
__device__ __forceinline__ v16bf make_afrag_f32(const float* row, int k0, int half) {
  Frag32B f;
  const float* p0 = row + k0 + half * 16;
  const float* p1 = p0 + 8;
#pragma unroll
  for (int i = 0; i < 8; ++i) f.s[i] = f2bf(p0[i]);
#pragma unroll
  for (int i = 0; i < 8; ++i) f.s[8 + i] = f2bf(p1[i]);
  return f.v;
}

// ---------------------------------------------------------------------------
// TDM: 2D tile load Global -> LDS via Tensor Data Mover (CDNA5 ISA ch.8).
// Descriptor bit packing per D# group tables (ISA p.152-156).
// This toolchain ships the 6-arg builtin:
//   (uint32x4 g0, int32x8 g1, int32x4 g2, int32x4 g3, int32x8 pad, i32 cpol)
// ---------------------------------------------------------------------------
#if __has_builtin(__builtin_amdgcn_tensor_load_to_lds) && \
    __has_builtin(__builtin_amdgcn_s_wait_tensorcnt)
#define USE_TDM 1
typedef unsigned int u32x4 __attribute__((ext_vector_type(4)));
typedef int          i32x8 __attribute__((ext_vector_type(8)));
typedef int          i32x4 __attribute__((ext_vector_type(4)));

__device__ __forceinline__ void tdm_load_2d_bf16(unsigned lds_off,
                                                 const void* gaddr,
                                                 unsigned tile_d0, unsigned tile_d1,
                                                 unsigned tensor_d0, unsigned tensor_d1,
                                                 unsigned long long stride0_elems)
{
  unsigned long long ga = (unsigned long long)(uintptr_t)gaddr;
  u32x4 g0;
  g0[0] = 1u;                                              // count=1, user mode
  g0[1] = lds_off;                                         // lds_addr
  g0[2] = (unsigned)(ga & 0xFFFFFFFFu);                    // global_addr[31:0]
  g0[3] = (unsigned)((ga >> 32) & 0x01FFFFFFu) | (2u << 30); // addr[56:32] | type=2
  i32x8 g1;
  g1[0] = (int)(1u << 16);                                 // data_size=1 (2 bytes)
  g1[1] = (int)((tensor_d0 & 0xFFFFu) << 16);              // tensor_dim0[15:0] @63:48
  g1[2] = (int)((tensor_d0 >> 16) | ((tensor_d1 & 0xFFFFu) << 16)); // @79:64,@95:80
  g1[3] = (int)((tensor_d1 >> 16) | (tile_d0 << 16));      // @111:96, tile_dim0 @127:112
  g1[4] = (int)(tile_d1 & 0xFFFFu);                        // tile_dim1 @143:128, tile_dim2=0
  g1[5] = (int)(unsigned)(stride0_elems & 0xFFFFFFFFu);    // dim0_stride @191:160
  g1[6] = (int)(unsigned)((stride0_elems >> 32) & 0xFFFFu);// dim0_stride hi @207:192
  g1[7] = 0;                                               // dim1_stride = 0 (2D)
  i32x4 z4 = {0, 0, 0, 0};
  i32x8 z8 = {0, 0, 0, 0, 0, 0, 0, 0};
  __builtin_amdgcn_tensor_load_to_lds(g0, g1, z4, z4, z8, 0);
}
#else
#define USE_TDM 0
#endif

// ---------------------------------------------------------------------------
// Weight convert: Wt[n*512 + k] = bf16(W[k*512 + n])   (transposed, bf16)
// ---------------------------------------------------------------------------
__global__ void __launch_bounds__(256)
convert_w(const float* __restrict__ W, unsigned short* __restrict__ Wt) {
  int i = blockIdx.x * 256 + threadIdx.x;     // i = k*512 + n, 262144 total
  int n = i & 511, k = i >> 9;
  Wt[n * DM + k] = f2bf(W[i]);
}

// ---------------------------------------------------------------------------
// GEMM: Out[M=8192, N=512] = A[M,512] @ W[512,512] + bias, times out_scale.
// Wt is bf16 transposed [N][K]. One wave -> 16x64 tile; 16 K-steps of 4 WMMAs.
// A_BF16: 0 = A is fp32, 1 = A is bf16(ushort)
// OUT_MODE: 0 = bf16 row-major, 1 = bf16 V-transposed [b][h][d][s], 2 = fp32
// ---------------------------------------------------------------------------
template<int A_BF16, int OUT_MODE>
__global__ void __launch_bounds__(128)
proj_gemm(const void* __restrict__ Aptr,
          const unsigned short* __restrict__ Wt,
          const float* __restrict__ bias,
          void* __restrict__ Out, float out_scale)
{
  const int lane = threadIdx.x & 31;
  const int l15  = lane & 15, half = lane >> 4;
  const int wid  = blockIdx.x * (blockDim.x >> 5) + (threadIdx.x >> 5);
  const int m0 = (wid >> 3) * 16;       // 512 row tiles
  const int n0 = (wid & 7) * 64;        // 8 col tiles of 64

  v8f acc[4];
#pragma unroll
  for (int f = 0; f < 4; ++f) acc[f] = zero8();

  const int arow = m0 + l15;

#pragma unroll 1
  for (int kk = 0; kk < 16; ++kk) {
    const int k0 = kk * 32;
    v16bf a;
    if (A_BF16) a = load_afrag((const unsigned short*)Aptr + arow * DM, k0, half);
    else        a = make_afrag_f32((const float*)Aptr + arow * DM, k0, half);
#pragma unroll
    for (int f = 0; f < 4; ++f) {
      const unsigned short* bp = Wt + (n0 + f * 16 + l15) * DM + k0 + 16 * half;
      v16bf b = *(const v16bf*)bp;
      acc[f] = __builtin_amdgcn_wmma_f32_16x16x32_bf16(false, a, false, b,
                                                       (short)0, acc[f], false, false);
    }
  }

#pragma unroll
  for (int f = 0; f < 4; ++f) {
    const int col = n0 + f * 16 + l15;
    const float bv = bias[col];
#pragma unroll
    for (int g = 0; g < 8; ++g) {
      const int row = m0 + half * 8 + g;
      float v = (acc[f][g] + bv) * out_scale;
      if (OUT_MODE == 0) {
        ((unsigned short*)Out)[row * DM + col] = f2bf(v);
      } else if (OUT_MODE == 1) {
        int b_ = row >> 12, s_ = row & 4095;     // token -> (batch, seq)
        int h_ = col >> 6,  d_ = col & 63;       // col   -> (head, d)
        ((unsigned short*)Out)[((b_ * NH + h_) * DHEAD + d_) * S_LEN + s_] = f2bf(v);
      } else {
        ((float*)Out)[row * DM + col] = v;
      }
    }
  }
}

// ---------------------------------------------------------------------------
// Flash attention. Block = 4 waves = 64 queries of one (b,h); each wave owns
// 16 queries. K/V staged 64 keys at a time into double-buffered LDS via TDM
// (each wave DMAs its quarter, one tile ahead of compute). Online softmax
// over 64-key passes; row-sum reduction deferred to the end.
// Qh,Kh: bf16 [B*S][512] (Q pre-scaled by 1/8). Vt: bf16 [b][h][d][s].
// ---------------------------------------------------------------------------
__global__ void __launch_bounds__(128)
attn_kernel(const unsigned short* __restrict__ Qh,
            const unsigned short* __restrict__ Kh,
            const unsigned short* __restrict__ Vt,
            unsigned short* __restrict__ Ctx)
{
  __shared__ __align__(16) unsigned short sK[2][64 * 64];  // [buf][key][d]
  __shared__ __align__(16) unsigned short sV[2][64 * 64];  // [buf][d][key]
  __shared__ __align__(16) unsigned short sP[4][16 * 64];  // per-wave P tile

  const int tid  = threadIdx.x;
  const int lane = tid & 31, w = tid >> 5;
  const int l15  = lane & 15, half = lane >> 4;

  const int bh = blockIdx.x >> 6;          // (b*8 + h), 0..15
  const int qt = blockIdx.x & 63;          // 64-query tile index
  const int b  = bh >> 3, h = bh & 7;

  const int qtok0 = b * S_LEN + qt * 64 + w * 16;   // this wave's first token

  v16bf qf[2];
  {
    const unsigned short* qrow = Qh + (qtok0 + l15) * DM + h * DHEAD;
    qf[0] = load_afrag(qrow, 0, half);     // d 0..31
    qf[1] = load_afrag(qrow, 32, half);    // d 32..63
  }

  v8f o[4];
#pragma unroll
  for (int f = 0; f < 4; ++f) o[f] = zero8();
  float mrow[8], lrow[8];                  // lrow: per-LANE partial sums
#pragma unroll
  for (int g = 0; g < 8; ++g) { mrow[g] = -3.0e38f; lrow[g] = 0.0f; }

  const unsigned short* Kbase = Kh + (size_t)b * S_LEN * DM + h * DHEAD;
  const unsigned short* Vbase = Vt + (size_t)bh * DHEAD * S_LEN;
  unsigned short* pw = sP[w];

#if USE_TDM
  const unsigned sK_off = (unsigned)(uintptr_t)(void*)&sK[0][0];
  const unsigned sV_off = (unsigned)(uintptr_t)(void*)&sV[0][0];
  // Each wave DMAs 16 key-rows of K and 16 d-rows of V (2KB each, rows of
  // 64 bf16 packed contiguously into LDS, matching sK/sV row-major layout).
  auto issue_tile = [&](int kt, unsigned bi) {
    tdm_load_2d_bf16(sK_off + bi * 8192u + (unsigned)w * 2048u,
                     Kbase + (size_t)(kt * 64 + w * 16) * DM,
                     /*tile*/ 64, 16, /*tensor*/ 64, 4096, /*stride*/ DM);
    tdm_load_2d_bf16(sV_off + bi * 8192u + (unsigned)w * 2048u,
                     Vbase + (size_t)(w * 16) * S_LEN + kt * 64,
                     /*tile*/ 64, 16, /*tensor*/ 64, 4096, /*stride*/ S_LEN);
  };
  issue_tile(0, 0u);                       // prologue: prefetch tile 0
#endif

#pragma unroll 1
  for (int kt = 0; kt < S_LEN / 64; ++kt) {
#if USE_TDM
    __builtin_amdgcn_s_wait_tensorcnt(0);  // my quarter of tile kt arrived
    __syncthreads();                       // everyone's quarter arrived AND
                                           // everyone finished tile kt-1
    if (kt + 1 < S_LEN / 64)
      issue_tile(kt + 1, (unsigned)((kt + 1) & 1));  // overlap DMA w/ compute
    const unsigned short* cK = &sK[kt & 1][0];
    const unsigned short* cV = &sV[kt & 1][0];
#else
    __syncthreads();
#pragma unroll
    for (int i = 0; i < 4; ++i) {
      int chunk = tid + i * 128;
      int r = chunk >> 3, c8 = (chunk & 7) * 8;
      *(uint4*)(&sK[0][0] + r * 64 + c8) =
          *(const uint4*)(Kbase + (size_t)(kt * 64 + r) * DM + c8);
    }
#pragma unroll
    for (int i = 0; i < 4; ++i) {
      int chunk = tid + i * 128;
      int d = chunk >> 3, c8 = (chunk & 7) * 8;
      *(uint4*)(&sV[0][0] + d * 64 + c8) =
          *(const uint4*)(Vbase + (size_t)d * S_LEN + kt * 64 + c8);
    }
    __syncthreads();
    const unsigned short* cK = &sK[0][0];
    const unsigned short* cV = &sV[0][0];
#endif

    // ---- scores: 16 queries x 64 keys (4 subtiles of 16) ----
    v8f sc[4];
#pragma unroll
    for (int sub = 0; sub < 4; ++sub) {
      v8f s = zero8();
      const unsigned short* kcol = cK + (sub * 16 + l15) * 64;
      v16bf b0 = *(const v16bf*)(kcol + 16 * half);        // d 0..31
      v16bf b1 = *(const v16bf*)(kcol + 32 + 16 * half);   // d 32..63
      s = __builtin_amdgcn_wmma_f32_16x16x32_bf16(false, qf[0], false, b0,
                                                  (short)0, s, false, false);
      s = __builtin_amdgcn_wmma_f32_16x16x32_bf16(false, qf[1], false, b1,
                                                  (short)0, s, false, false);
      sc[sub] = s;
    }

    // ---- online softmax over the 64-key pass ----
#pragma unroll
    for (int g = 0; g < 8; ++g) {
      float t = fmaxf(fmaxf(sc[0][g], sc[1][g]), fmaxf(sc[2][g], sc[3][g]));
      t = fmaxf(t, __shfl_xor(t, 1, 32));
      t = fmaxf(t, __shfl_xor(t, 2, 32));
      t = fmaxf(t, __shfl_xor(t, 4, 32));
      t = fmaxf(t, __shfl_xor(t, 8, 32));        // row-uniform tile max
      float mnew  = fmaxf(mrow[g], t);
      float alpha = __expf(mrow[g] - mnew);
      mrow[g] = mnew;
      o[0][g] *= alpha; o[1][g] *= alpha; o[2][g] *= alpha; o[3][g] *= alpha;
      int r = half * 8 + g;
      float psum = 0.0f;
#pragma unroll
      for (int sub = 0; sub < 4; ++sub) {
        float p = __expf(sc[sub][g] - mnew);
        psum += p;
        pw[r * 64 + sub * 16 + l15] = f2bf(p);
      }
      lrow[g] = lrow[g] * alpha + psum;          // per-lane partial row sum
    }

    // ---- PV: 16x64 += P(16x64) @ V(64x64), two 32-key chunks ----
#pragma unroll
    for (int kc = 0; kc < 2; ++kc) {
      v16bf pa = load_afrag(pw + l15 * 64, kc * 32, half);  // wave-internal LDS RAW
#pragma unroll
      for (int f = 0; f < 4; ++f) {
        v16bf bv = *(const v16bf*)(cV + (f * 16 + l15) * 64 + kc * 32 + 16 * half);
        o[f] = __builtin_amdgcn_wmma_f32_16x16x32_bf16(false, pa, false, bv,
                                                       (short)0, o[f], false, false);
      }
    }
  }

  // ---- deferred row-sum reduction (once), normalize, write context ----
#pragma unroll
  for (int g = 0; g < 8; ++g) {
    float ts = lrow[g];
    ts += __shfl_xor(ts, 1, 32);
    ts += __shfl_xor(ts, 2, 32);
    ts += __shfl_xor(ts, 4, 32);
    ts += __shfl_xor(ts, 8, 32);
    lrow[g] = 1.0f / ts;
  }
#pragma unroll
  for (int f = 0; f < 4; ++f) {
#pragma unroll
    for (int g = 0; g < 8; ++g) {
      int tok = qtok0 + half * 8 + g;
      int col = h * DHEAD + f * 16 + l15;
      Ctx[(size_t)tok * DM + col] = f2bf(o[f][g] * lrow[g]);
    }
  }
}

// ---------------------------------------------------------------------------
// Workspace layout (bytes): Qh 0..8M, Kh 8..16M, Vt 16..24M, Ctx 24..32M,
// weights bf16-T 32..34M. Requires ws_size >= 34 MB.
// ---------------------------------------------------------------------------
extern "C" void kernel_launch(void* const* d_in, const int* in_sizes, int n_in,
                              void* d_out, int out_size, void* d_ws, size_t ws_size,
                              hipStream_t stream)
{
  const float* q  = (const float*)d_in[0];
  const float* k  = (const float*)d_in[1];
  const float* v  = (const float*)d_in[2];
  const float* Wq = (const float*)d_in[3];
  const float* bq = (const float*)d_in[4];
  const float* Wk = (const float*)d_in[5];
  const float* bk = (const float*)d_in[6];
  const float* Wv = (const float*)d_in[7];
  const float* bv = (const float*)d_in[8];
  const float* Wo = (const float*)d_in[9];
  const float* bo = (const float*)d_in[10];

  char* ws = (char*)d_ws;
  unsigned short* Qh  = (unsigned short*)(ws);
  unsigned short* Kh  = (unsigned short*)(ws + (size_t)(8u << 20));
  unsigned short* Vt  = (unsigned short*)(ws + (size_t)(16u << 20));
  unsigned short* Ctx = (unsigned short*)(ws + (size_t)(24u << 20));
  unsigned short* Wqt = (unsigned short*)(ws + (size_t)(32u << 20));
  unsigned short* Wkt = Wqt + 512 * 512;
  unsigned short* Wvt = Wkt + 512 * 512;
  unsigned short* Wot = Wvt + 512 * 512;

  convert_w<<<1024, 256, 0, stream>>>(Wq, Wqt);
  convert_w<<<1024, 256, 0, stream>>>(Wk, Wkt);
  convert_w<<<1024, 256, 0, stream>>>(Wv, Wvt);
  convert_w<<<1024, 256, 0, stream>>>(Wo, Wot);

  // Q projection pre-scaled by 1/sqrt(d_k) = 0.125
  proj_gemm<0, 0><<<1024, 128, 0, stream>>>(q, Wqt, bq, Qh, 0.125f);
  proj_gemm<0, 0><<<1024, 128, 0, stream>>>(k, Wkt, bk, Kh, 1.0f);
  proj_gemm<0, 1><<<1024, 128, 0, stream>>>(v, Wvt, bv, Vt, 1.0f);

  attn_kernel<<<1024, 128, 0, stream>>>(Qh, Kh, Vt, Ctx);

  proj_gemm<1, 2><<<1024, 128, 0, stream>>>(Ctx, Wot, bo, d_out, 1.0f);
}
